// BrainDecoder_6957847020193
// MI455X (gfx1250) — matrix operand order repaired and checked by reading, not compiled, not run on hardware
//
#include <hip/hip_runtime.h>

typedef __attribute__((ext_vector_type(16))) _Float16 v16h;
typedef __attribute__((ext_vector_type(8)))  float    v8f;
typedef __attribute__((ext_vector_type(4)))  unsigned int v4u;
typedef __attribute__((ext_vector_type(8)))  unsigned int v8u;

#define HIDDEN 128
#define NNODES 400
#define NTPAD  416          // padded node stride for transposed h (16-half aligned)
#define NBATCH 8
#define NTIME  32
#define BT     256          // NBATCH*NTIME
#define ROWS   102400       // BT*NNODES
#define GSZ    160000       // NNODES*NNODES

// ---------------------------------------------------------------- utilities
__global__ void k_zero(float* p, int n) {
  int i = blockIdx.x * blockDim.x + threadIdx.x;
  if (i < n) p[i] = 0.f;
}

__global__ void k_cvt(const float* s, _Float16* d, int n) {
  int i = blockIdx.x * blockDim.x + threadIdx.x;
  if (i < n) d[i] = (_Float16)s[i];
}

// ------------------------------------------------- per-graph 70th percentile
// two-level 1024-bin histogram radix-select on uniform [0,1) values.
__global__ void k_thr(const float* __restrict__ dyn_a, float* __restrict__ thr) {
  int bt = blockIdx.x;
  int tid = threadIdx.x;
  const float* a = dyn_a + (size_t)bt * GSZ;
  __shared__ int hist[1024];
  __shared__ float sres[4];

  for (int i = tid; i < 1024; i += 256) hist[i] = 0;
  __syncthreads();
  for (int i = tid; i < GSZ; i += 256) {
    float v = a[i];
    int b = (int)(v * 1024.f); b = min(max(b, 0), 1023);
    atomicAdd(&hist[b], 1);
  }
  __syncthreads();

  float qf = 0.7f * (float)(GSZ - 1);
  int   k  = (int)qf;
  float frac = qf - (float)k;

  if (tid == 0) {
    int cum = 0, b0 = -1, b1 = -1, base0 = 0, base1 = 0;
    for (int i = 0; i < 1024; i++) {
      int c = hist[i];
      if (b0 < 0 && cum + c > k)     { b0 = i; base0 = cum; }
      if (b1 < 0 && cum + c > k + 1) { b1 = i; base1 = cum; }
      cum += c;
      if (b0 >= 0 && b1 >= 0) break;
    }
    sres[0] = (float)b0; sres[1] = (float)base0;
    sres[2] = (float)b1; sres[3] = (float)base1;
  }
  __syncthreads();
  int tb[2]    = { (int)sres[0], (int)sres[2] };
  int tbase[2] = { (int)sres[1], (int)sres[3] };
  int ranks[2] = { k, k + 1 };

  for (int p = 0; p < 2; p++) {
    __syncthreads();
    for (int i = tid; i < 1024; i += 256) hist[i] = 0;
    __syncthreads();
    float lo = tb[p] * (1.f / 1024.f);
    for (int i = tid; i < GSZ; i += 256) {
      float v = a[i];
      int b = (int)(v * 1024.f); b = min(max(b, 0), 1023);
      if (b == tb[p]) {
        int s = (int)((v - lo) * (1024.f * 1024.f));
        s = min(max(s, 0), 1023);
        atomicAdd(&hist[s], 1);
      }
    }
    __syncthreads();
    if (tid == 0) {
      int cum = tbase[p], sb = 1023;
      for (int i = 0; i < 1024; i++) {
        if (cum + hist[i] > ranks[p]) { sb = i; break; }
        cum += hist[i];
      }
      sres[p] = lo + (sb + 0.5f) * (1.f / (1024.f * 1024.f));
    }
  }
  __syncthreads();
  if (tid == 0) thr[bt] = sres[0] + frac * (sres[1] - sres[0]);
}

// ------------------------------------------ initial linear: window(50) -> 128
__global__ void k_init(const float* __restrict__ dyn_t, const float* __restrict__ w,
                       const float* __restrict__ bias, _Float16* __restrict__ h,
                       _Float16* __restrict__ hT) {
  int row = blockIdx.x;
  int c = threadIdx.x;
  __shared__ float xr[50];
  if (c < 50) xr[c] = dyn_t[(size_t)row * 50 + c];
  __syncthreads();
  float acc = bias[c];
  const float* wr = w + (size_t)c * 50;
  for (int kk = 0; kk < 50; kk++) acc += xr[kk] * wr[kk];
  _Float16 hv = (_Float16)acc;
  h[(size_t)row * HIDDEN + c] = hv;
  int bt = row / NNODES, n = row - bt * NNODES;
  hT[((size_t)bt * HIDDEN + c) * NTPAD + n] = hv;
}

// ----------------------------- agg = binarize(dyn_a) @ h + eps*h   (f16 WMMA)
// adjacency tile DMA'd into LDS by the Tensor Data Mover, binarized in LDS,
// B operand streamed from the transposed h copy (32B-aligned v16h loads).
__global__ void k_agg(const float* __restrict__ dyn_a, const float* __restrict__ thrp,
                      const float* __restrict__ epsp, const _Float16* __restrict__ hin,
                      const _Float16* __restrict__ hT, _Float16* __restrict__ out) {
  int bt = blockIdx.y;
  int i0 = blockIdx.x * 16;
  int tid = threadIdx.x;

  __shared__ float    Araw[16][NNODES];     // raw adjacency tile (TDM destination)
  __shared__ _Float16 As[16][NTPAD];        // binarized f16 tile, K-padded

  int lane = tid & 31, wv = tid >> 5;

  // ---- TDM: DMA 16x400 f32 tile of dyn_a into LDS (one wave issues it)
  if (wv == 0) {
    unsigned long long ga =
        (unsigned long long)(uintptr_t)(dyn_a + (size_t)bt * GSZ + (size_t)i0 * NNODES);
    unsigned lds_off = (unsigned)(uintptr_t)&Araw[0][0];   // generic addr[31:0] = LDS offset
    v4u g0;
    g0[0] = 1u;                                            // count=1, user descriptor
    g0[1] = lds_off;                                       // lds_addr
    g0[2] = (unsigned)ga;                                  // global_addr[31:0]
    g0[3] = (unsigned)((ga >> 32) & 0x01FFFFFFu) | (2u << 30);  // addr[56:32] | type=2
    v8u g1;
    g1[0] = 0x00020000u;                 // workgroup_mask=0, data_size=2 (4B)
    g1[1] = (unsigned)NNODES << 16;      // tensor_dim0 = 400 (lo16 at bit48)
    g1[2] = 16u << 16;                   // tensor_dim0 hi=0 | tensor_dim1 = 16
    g1[3] = (unsigned)NNODES << 16;      // tensor_dim1 hi=0 | tile_dim0 = 400
    g1[4] = 16u;                         // tile_dim1 = 16, tile_dim2 = 0
    g1[5] = (unsigned)NNODES;            // tensor_dim0_stride = 400
    g1[6] = 0u;
    g1[7] = 0u;
    asm volatile("tensor_load_to_lds %0, %1" :: "s"(g0), "s"(g1) : "memory");
    __builtin_amdgcn_s_wait_tensorcnt(0);
  }
  __syncthreads();

  // ---- binarize in LDS (f32 -> {0,1} f16), zero K-pad columns
  float th = thrp[bt];
  for (int i = tid; i < 16 * NNODES; i += 256) {
    int rr = i / NNODES, cc = i - rr * NNODES;
    As[rr][cc] = (Araw[rr][cc] > th) ? (_Float16)1.0f : (_Float16)0.0f;
  }
  { int rr = tid >> 4, cc = NNODES + (tid & 15); As[rr][cc] = (_Float16)0.0f; }
  __syncthreads();

  int m = lane & 15, g = lane >> 4;
  int c = wv * 16 + (lane & 15);
  const _Float16* hTc = hT + ((size_t)bt * HIDDEN + c) * NTPAD;
  v8f acc = {};
  for (int ks = 0; ks < 13; ks++) {
    int k0 = ks * 32;
    v16h av;
#pragma unroll
    for (int j = 0; j < 8; j++) {
      av[j]     = As[m][k0 + g * 16 + j];
      av[j + 8] = As[m][k0 + g * 16 + 8 + j];
    }
    v16h bv = *reinterpret_cast<const v16h*>(hTc + k0 + g * 16);
    acc = __builtin_amdgcn_wmma_f32_16x16x32_f16(false, av, false, bv,
                                                 (short)0, acc, false, false);
  }

  float eps = epsp[0];
#pragma unroll
  for (int r = 0; r < 8; r++) {
    int grow = i0 + r + g * 8;
    size_t o = ((size_t)bt * NNODES + grow) * HIDDEN + c;
    out[o] = (_Float16)(acc[r] + eps * (float)hin[o]);
  }
}

// --------------------- fused  Y = relu(LN(X @ W^T + b))   (f16 WMMA, in-place ok)
// optionally writes a transposed copy of the output (for the next agg / pool).
__global__ void k_lin(const _Float16* __restrict__ xin, const _Float16* __restrict__ W,
                      const float* __restrict__ bias, const float* __restrict__ gamma,
                      const float* __restrict__ beta, _Float16* __restrict__ out,
                      _Float16* __restrict__ outT) {
  int gr0 = blockIdx.x * 16;
  int tid = threadIdx.x;
  __shared__ alignas(16) _Float16 Xs[16][HIDDEN];
  __shared__ float Ys[16][HIDDEN];
  __shared__ float redS[256], redQ[256];

  ((uint4*)Xs)[tid] = ((const uint4*)(xin + (size_t)gr0 * HIDDEN))[tid];
  __syncthreads();

  int lane = tid & 31, wv = tid >> 5;
  int m = lane & 15, g = lane >> 4;
  int c = wv * 16 + (lane & 15);
  v8f acc = {};
#pragma unroll
  for (int ks = 0; ks < 4; ks++) {
    int k0 = ks * 32;
    v16h av;
#pragma unroll
    for (int j = 0; j < 8; j++) {
      av[j]     = Xs[m][k0 + g * 16 + j];
      av[j + 8] = Xs[m][k0 + g * 16 + 8 + j];
    }
    v16h bv = *reinterpret_cast<const v16h*>(W + (size_t)c * HIDDEN + k0 + g * 16);
    acc = __builtin_amdgcn_wmma_f32_16x16x32_f16(false, av, false, bv,
                                                 (short)0, acc, false, false);
  }
#pragma unroll
  for (int r = 0; r < 8; r++) Ys[r + g * 8][c] = acc[r] + bias[c];
  __syncthreads();

  int mm = tid >> 4, u = tid & 15;
  float s = 0.f, q2 = 0.f;
#pragma unroll
  for (int j = 0; j < 8; j++) { float x = Ys[mm][u + 16 * j]; s += x; q2 += x * x; }
  redS[tid] = s; redQ[tid] = q2; __syncthreads();
  for (int o = 8; o; o >>= 1) {
    if (u < o) { redS[tid] += redS[tid + o]; redQ[tid] += redQ[tid + o]; }
    __syncthreads();
  }
  float mean = redS[mm * 16] * (1.f / 128.f);
  float var  = redQ[mm * 16] * (1.f / 128.f) - mean * mean;
  float rs   = rsqrtf(var + 1e-5f);
  int grow = gr0 + mm;
  int btg = grow / NNODES, nloc = grow - btg * NNODES;
#pragma unroll
  for (int j = 0; j < 8; j++) {
    int cc = u + 16 * j;
    float y = fmaxf((Ys[mm][cc] - mean) * rs * gamma[cc] + beta[cc], 0.f);
    _Float16 yh = (_Float16)y;
    out[(size_t)grow * HIDDEN + cc] = yh;
    if (outT) outT[((size_t)btg * HIDDEN + cc) * NTPAD + nloc] = yh;
  }
}

// ------------------- attention pooling over nodes -> X_dec[T,B,C]
__global__ void k_pool(const _Float16* __restrict__ h, const _Float16* __restrict__ hT,
                       const float* __restrict__ e_s, const float* __restrict__ X_enc,
                       const int* __restrict__ samp, float* __restrict__ X_dec) {
  int bt = blockIdx.x;
  int b = bt >> 5, t = bt & 31;
  int tid = threadIdx.x;
  __shared__ float qv[HIDDEN];
  __shared__ float sc[NNODES];
  __shared__ float red[HIDDEN];

  int ep = samp[t] - 1;
  qv[tid] = e_s[b * HIDDEN + tid] + X_enc[((size_t)ep * NBATCH + b) * HIDDEN + tid];
  __syncthreads();

  const _Float16* hb = h + (size_t)bt * NNODES * HIDDEN;
  for (int n = tid; n < NNODES; n += 128) {
    const _Float16* hr = hb + (size_t)n * HIDDEN;
    float s = 0.f;
    for (int cc = 0; cc < HIDDEN; cc++) s += (float)hr[cc] * qv[cc];
    sc[n] = s;
  }
  __syncthreads();
  float mx = -1e30f;
  for (int n = tid; n < NNODES; n += 128) mx = fmaxf(mx, sc[n]);
  red[tid] = mx; __syncthreads();
  for (int o = 64; o; o >>= 1) { if (tid < o) red[tid] = fmaxf(red[tid], red[tid + o]); __syncthreads(); }
  mx = red[0]; __syncthreads();
  float se = 0.f;
  for (int n = tid; n < NNODES; n += 128) se += __expf(sc[n] - mx);
  red[tid] = se; __syncthreads();
  for (int o = 64; o; o >>= 1) { if (tid < o) red[tid] += red[tid + o]; __syncthreads(); }
  float inv = 1.f / red[0]; __syncthreads();
  for (int n = tid; n < NNODES; n += 128) sc[n] = __expf(sc[n] - mx) * inv + 1.f;
  __syncthreads();
  const _Float16* hc = hT + ((size_t)bt * HIDDEN + tid) * NTPAD;
  float acc = 0.f;
  for (int n = 0; n < NNODES; n++) acc += (float)hc[n] * sc[n];
  X_dec[((size_t)t * NBATCH + b) * HIDDEN + tid] = acc;
}

// ------------------- qkv projections (q from X_dec, k/v from X_enc)
__global__ void k_qkv(const float* __restrict__ X_dec, const float* __restrict__ X_enc,
                      const float* __restrict__ wqkv, const float* __restrict__ bqkv,
                      float* __restrict__ qp, float* __restrict__ kp, float* __restrict__ vp) {
  int r = blockIdx.x;
  int c = threadIdx.x;
  __shared__ float xd[HIDDEN], xe[HIDDEN];
  xd[c] = X_dec[(size_t)r * HIDDEN + c];
  xe[c] = X_enc[(size_t)r * HIDDEN + c];
  __syncthreads();
  float aq = bqkv[c], ak = bqkv[HIDDEN + c], av = bqkv[2 * HIDDEN + c];
  for (int kk = 0; kk < HIDDEN; kk++) {
    aq += xd[kk] * wqkv[(size_t)c * HIDDEN + kk];
    ak += xe[kk] * wqkv[(size_t)(HIDDEN + c) * HIDDEN + kk];
    av += xe[kk] * wqkv[(size_t)(2 * HIDDEN + c) * HIDDEN + kk];
  }
  qp[(size_t)r * HIDDEN + c] = aq;
  kp[(size_t)r * HIDDEN + c] = ak;
  vp[(size_t)r * HIDDEN + c] = av;
}

// ------------------- per (b,head,t) attention row; 32 lanes = S
__global__ void k_attn(const float* __restrict__ qp, const float* __restrict__ kp,
                       const float* __restrict__ vp, float* __restrict__ attw,
                       float* __restrict__ heads) {
  int idx = blockIdx.x;                 // (b*4 + h)*32 + t
  int t = idx & 31, hh = (idx >> 5) & 3, b = idx >> 7;
  int s = threadIdx.x;
  __shared__ float wsh[32];

  float scv = 0.f;
  for (int d = 0; d < 32; d++)
    scv += qp[((size_t)t * NBATCH + b) * HIDDEN + hh * 32 + d] *
           kp[((size_t)s * NBATCH + b) * HIDDEN + hh * 32 + d];
  scv *= 0.17677669529663687f;          // 1/sqrt(32)

  float mx = scv;
  for (int o = 16; o; o >>= 1) mx = fmaxf(mx, __shfl_xor(mx, o));
  float e = __expf(scv - mx);
  float sum = e;
  for (int o = 16; o; o >>= 1) sum += __shfl_xor(sum, o);
  float w = e / sum;
  attw[(size_t)idx * 32 + s] = w;
  wsh[s] = w;
  __syncthreads();
  int dd = threadIdx.x;
  float oacc = 0.f;
  for (int s2 = 0; s2 < 32; s2++)
    oacc += wsh[s2] * vp[((size_t)s2 * NBATCH + b) * HIDDEN + hh * 32 + dd];
  heads[((size_t)t * NBATCH + b) * HIDDEN + hh * 32 + dd] = oacc;
}

// mean over heads -> tatt slice for this layer
__global__ void k_attnred(const float* __restrict__ attw, float* __restrict__ outp, int L) {
  int i = blockIdx.x * 256 + threadIdx.x;          // B*T*S = 8192
  if (i >= 8192) return;
  int b = i >> 10, t = (i >> 5) & 31, s = i & 31;
  float v = 0.f;
  for (int hh = 0; hh < 4; hh++)
    v += attw[(((size_t)(b * 4 + hh) * 32) + t) * 32 + s];
  outp[16 + (((size_t)(b * 2 + L) * 32) + t) * 32 + s] = v * 0.25f;
}

// ------------------- output projection of MHA
__global__ void k_oproj(const float* __restrict__ heads, const float* __restrict__ wo,
                        const float* __restrict__ bo, float* __restrict__ att) {
  int r = blockIdx.x;
  int c = threadIdx.x;
  __shared__ float x[HIDDEN];
  x[c] = heads[(size_t)r * HIDDEN + c];
  __syncthreads();
  float a = bo[c];
  for (int kk = 0; kk < HIDDEN; kk++) a += x[kk] * wo[(size_t)c * HIDDEN + kk];
  att[(size_t)r * HIDDEN + c] = a;
}

// ------------------- LN -> FFN -> residual -> LN
__global__ void k_post(const float* __restrict__ att,
                       const float* __restrict__ g1, const float* __restrict__ b1g,
                       const float* __restrict__ g2, const float* __restrict__ b2g,
                       const float* __restrict__ w1, const float* __restrict__ bb1,
                       const float* __restrict__ w2, const float* __restrict__ bb2,
                       float* __restrict__ xout) {
  int r = blockIdx.x;
  int tid = threadIdx.x;
  __shared__ float x[HIDDEN], x2a[2 * HIDDEN], red[HIDDEN], red2[HIDDEN];

  float v = att[(size_t)r * HIDDEN + tid];
  red[tid] = v; red2[tid] = v * v; __syncthreads();
  for (int o = 64; o; o >>= 1) { if (tid < o) { red[tid] += red[tid + o]; red2[tid] += red2[tid + o]; } __syncthreads(); }
  float mean = red[0] * (1.f / 128.f);
  float var  = red2[0] * (1.f / 128.f) - mean * mean;
  float xv = (v - mean) * rsqrtf(var + 1e-5f) * g1[tid] + b1g[tid];
  __syncthreads();
  x[tid] = xv;
  __syncthreads();
  for (int j = 0; j < 2; j++) {
    int o = tid + HIDDEN * j;
    float a = bb1[o];
    for (int kk = 0; kk < HIDDEN; kk++) a += x[kk] * w1[(size_t)o * HIDDEN + kk];
    x2a[o] = fmaxf(a, 0.f);
  }
  __syncthreads();
  float a2 = bb2[tid];
  for (int o = 0; o < 2 * HIDDEN; o++) a2 += x2a[o] * w2[(size_t)tid * 2 * HIDDEN + o];
  float xr = xv + a2;
  __syncthreads();
  red[tid] = xr; red2[tid] = xr * xr; __syncthreads();
  for (int o = 64; o; o >>= 1) { if (tid < o) { red[tid] += red[tid + o]; red2[tid] += red2[tid + o]; } __syncthreads(); }
  mean = red[0] * (1.f / 128.f);
  var  = red2[0] * (1.f / 128.f) - mean * mean;
  xout[(size_t)r * HIDDEN + tid] = (xr - mean) * rsqrtf(var + 1e-5f) * g2[tid] + b2g[tid];
}

__global__ void k_hdyn(const float* __restrict__ xout, float* __restrict__ h_dyn) {
  int b = blockIdx.x, c = threadIdx.x;
  float s = 0.f;
  for (int t = 0; t < NTIME; t++) s += xout[((size_t)t * NBATCH + b) * HIDDEN + c];
  h_dyn[b * HIDDEN + c] = s;
}

__global__ void k_logit(const float* __restrict__ h_dyn, const float* __restrict__ out_w,
                        const float* __restrict__ out_b, float* __restrict__ d_out) {
  int tid = threadIdx.x;
  if (tid < 16) {
    int b = tid >> 1, cl = tid & 1;
    float v = out_b[cl];
    for (int c = 0; c < HIDDEN; c++) v += h_dyn[b * HIDDEN + c] * out_w[cl * HIDDEN + c];
    d_out[b * 2 + cl] += v;
  }
}

// ---------------------------------------------------------------- launch
extern "C" void kernel_launch(void* const* d_in, const int* in_sizes, int n_in,
                              void* d_out, int out_size, void* d_ws, size_t ws_size,
                              hipStream_t stream) {
  (void)in_sizes; (void)n_in; (void)out_size; (void)ws_size;

  const float* dyn_t  = (const float*)d_in[0];
  const float* dyn_a  = (const float*)d_in[1];
  const float* e_s    = (const float*)d_in[2];
  const float* X_enc  = (const float*)d_in[3];
  const int*   samp   = (const int*)d_in[4];
  const float* init_w = (const float*)d_in[5];
  const float* init_b = (const float*)d_in[6];
  auto LIN = [&](int L, int j) -> const float* { return (const float*)d_in[7 + L * 23 + j]; };
  // layer member order (dict insertion): 0 eps, 1 g_w1, 2 g_b1, 3 g_ln1_g, 4 g_ln1_b,
  // 5 g_w2, 6 g_b2, 7 g_ln2_g, 8 g_ln2_b, 9 attn_wqkv, 10 attn_bqkv, 11 attn_wo,
  // 12 attn_bo, 13 t_ln1_g, 14 t_ln1_b, 15 t_ln2_g, 16 t_ln2_b, 17 t_w1, 18 t_b1,
  // 19 t_w2, 20 t_b2, 21 out_w, 22 out_b

  char* ws = (char*)d_ws;
  size_t off = 0;
  auto alloc = [&](size_t bytes) -> char* {
    char* p = ws + off;
    off = (off + bytes + 255) & ~(size_t)255;
    return p;
  };
  float*    thr   = (float*)alloc(BT * sizeof(float));
  _Float16* hA    = (_Float16*)alloc((size_t)ROWS * HIDDEN * 2);
  _Float16* hB    = (_Float16*)alloc((size_t)ROWS * HIDDEN * 2);
  _Float16* hT    = (_Float16*)alloc((size_t)BT * HIDDEN * NTPAD * 2);
  _Float16* w16   = (_Float16*)alloc((size_t)4 * HIDDEN * HIDDEN * 2);
  float*    X_dec = (float*)alloc((size_t)NTIME * NBATCH * HIDDEN * 4);
  float*    qp    = (float*)alloc((size_t)NTIME * NBATCH * HIDDEN * 4);
  float*    kp    = (float*)alloc((size_t)NTIME * NBATCH * HIDDEN * 4);
  float*    vp    = (float*)alloc((size_t)NTIME * NBATCH * HIDDEN * 4);
  float*    heads = (float*)alloc((size_t)NTIME * NBATCH * HIDDEN * 4);
  float*    attw  = (float*)alloc((size_t)NBATCH * 4 * NTIME * NTIME * 4);
  float*    att   = (float*)alloc((size_t)NTIME * NBATCH * HIDDEN * 4);
  float*    xout  = (float*)alloc((size_t)NTIME * NBATCH * HIDDEN * 4);
  float*    h_dyn = (float*)alloc((size_t)NBATCH * HIDDEN * 4);

  // weight conversion (g_w1/g_w2 per layer), zero output + hT pad region
  for (int L = 0; L < 2; L++) {
    k_cvt<<<64, 256, 0, stream>>>(LIN(L, 1), w16 + (size_t)(L * 2 + 0) * HIDDEN * HIDDEN, HIDDEN * HIDDEN);
    k_cvt<<<64, 256, 0, stream>>>(LIN(L, 5), w16 + (size_t)(L * 2 + 1) * HIDDEN * HIDDEN, HIDDEN * HIDDEN);
  }
  k_zero<<<17, 256, 0, stream>>>((float*)d_out, 16 + NBATCH * 2 * NTIME * NTIME);
  {
    int hTfloats = BT * HIDDEN * NTPAD / 2;   // zero whole hT once (covers K-pad tail)
    k_zero<<<(hTfloats + 255) / 256, 256, 0, stream>>>((float*)hT, hTfloats);
  }

  k_thr<<<BT, 256, 0, stream>>>(dyn_a, thr);
  k_init<<<ROWS, 128, 0, stream>>>(dyn_t, init_w, init_b, hA, hT);

  _Float16* cur = hA;
  _Float16* oth = hB;
  for (int L = 0; L < 2; L++) {
    k_agg<<<dim3(NNODES / 16, BT), 256, 0, stream>>>(dyn_a, thr, LIN(L, 0), cur, hT, oth);
    k_lin<<<ROWS / 16, 256, 0, stream>>>(oth, w16 + (size_t)(L * 2 + 0) * HIDDEN * HIDDEN,
                                         LIN(L, 2), LIN(L, 3), LIN(L, 4), oth, (_Float16*)nullptr);
    k_lin<<<ROWS / 16, 256, 0, stream>>>(oth, w16 + (size_t)(L * 2 + 1) * HIDDEN * HIDDEN,
                                         LIN(L, 6), LIN(L, 7), LIN(L, 8), oth, hT);
    k_pool<<<BT, 128, 0, stream>>>(oth, hT, e_s, X_enc, samp, X_dec);
    k_qkv<<<NTIME * NBATCH, 128, 0, stream>>>(X_dec, X_enc, LIN(L, 9), LIN(L, 10), qp, kp, vp);
    k_attn<<<NBATCH * 4 * NTIME, 32, 0, stream>>>(qp, kp, vp, attw, heads);
    k_attnred<<<32, 256, 0, stream>>>(attw, (float*)d_out, L);
    k_oproj<<<NTIME * NBATCH, 128, 0, stream>>>(heads, LIN(L, 11), LIN(L, 12), att);
    k_post<<<NTIME * NBATCH, 128, 0, stream>>>(att, LIN(L, 13), LIN(L, 14), LIN(L, 15), LIN(L, 16),
                                               LIN(L, 17), LIN(L, 18), LIN(L, 19), LIN(L, 20), xout);
    k_hdyn<<<NBATCH, 128, 0, stream>>>(xout, h_dyn);
    k_logit<<<1, 32, 0, stream>>>(h_dyn, LIN(L, 21), LIN(L, 22), (float*)d_out);
    _Float16* tmp = cur; cur = oth; oth = tmp;
  }
}
